// torch_model_47837345743308
// MI455X (gfx1250) — matrix operand order repaired and checked by reading, not compiled, not run on hardware
//
#include <hip/hip_runtime.h>
#include <hip/hip_bf16.h>
#include <math.h>

// ---------------- constants ----------------
#define BB 8
#define SS 256
#define HH 768
#define NLAY 12
#define NHEAD 12
#define FFD 3072
#define NC 9
#define DHD 64
#define MTOK (BB * SS)   // 2048

// ---------------- types ----------------
typedef __bf16 bf16_t;
typedef bf16_t v16bf __attribute__((ext_vector_type(16)));
typedef float  v8f   __attribute__((ext_vector_type(8)));
typedef unsigned int u32x4 __attribute__((ext_vector_type(4)));

union Frag {
  u32x4 q[2];
  v16bf v;
};

__device__ __forceinline__ unsigned short f2bf(float f) {
  unsigned u = __float_as_uint(f);
  u += 0x7FFFu + ((u >> 16) & 1u);       // round-to-nearest-even
  return (unsigned short)(u >> 16);
}

__device__ __forceinline__ float gelu_tanh(float x) {
  float x3 = x * x * x;
  return 0.5f * x * (1.0f + tanhf(0.7978845608028654f * (x + 0.044715f * x3)));
}

// ---------------- weight convert + transpose: W[K][N] f32 -> Wt[N][K] bf16 ----------------
__global__ __launch_bounds__(256)
void convT_kernel(const float* __restrict__ W, unsigned short* __restrict__ Wt, int K, int N) {
  long long z  = blockIdx.y;
  long long KN = (long long)K * N;
  const float* w = W + z * KN;
  unsigned short* wt = Wt + z * KN;
  long long idx = (long long)blockIdx.x * 256 + threadIdx.x;
  if (idx >= KN) return;
  int n = (int)(idx / K), k = (int)(idx % K);
  wt[(long long)n * K + k] = f2bf(w[(long long)k * N + n]);
}

// ---------------- batched WMMA GEMM: C[M,N] = A[M,K] * Bt[N,K]^T (+bias, +gelu) ----------------
// batch z decomposed as zb = z / bh, zh = z % bh; pointer offsets via per-factor strides.
__global__ __launch_bounds__(128)
void wmma_gemm_bf16(const unsigned short* __restrict__ A, int lda, long long sAb, long long sAh,
                    const unsigned short* __restrict__ Bt, int ldb, long long sBb, long long sBh,
                    float* __restrict__ C, unsigned short* __restrict__ Obf,
                    int ldc, long long sCb, long long sCh,
                    const float* __restrict__ bias,
                    int M, int N, int K, int bh, int act)
{
  int z = blockIdx.z;
  int zb = z / bh, zh = z % bh;
  A  += (long long)zb * sAb + (long long)zh * sAh;
  Bt += (long long)zb * sBb + (long long)zh * sBh;
  long long coff = (long long)zb * sCb + (long long)zh * sCh;

  int wave = threadIdx.x >> 5;
  int lane = threadIdx.x & 31;
  int lh = lane >> 4, lm = lane & 15;
  int m0 = blockIdx.x * 64 + (wave >> 1) * 32;
  int n0 = blockIdx.y * 64 + (wave & 1) * 32;

  v8f zero = {0.f, 0.f, 0.f, 0.f, 0.f, 0.f, 0.f, 0.f};
  v8f acc[2][2];
  acc[0][0] = zero; acc[0][1] = zero; acc[1][0] = zero; acc[1][1] = zero;

  for (int k0 = 0; k0 < K; k0 += 32) {
    Frag a[2], bq[2];
#pragma unroll
    for (int mt = 0; mt < 2; mt++) {
      int row = m0 + mt * 16 + lm; if (row >= M) row = M - 1;
      const u32x4* p = (const u32x4*)(A + (size_t)row * lda + k0);
      a[mt].q[0] = p[lh];       // K = k0 + lh*8 + {0..7}
      a[mt].q[1] = p[2 + lh];   // K = k0 + 16 + lh*8 + {0..7}
    }
#pragma unroll
    for (int nt = 0; nt < 2; nt++) {
      int col = n0 + nt * 16 + lm; if (col >= N) col = N - 1;
      const u32x4* p = (const u32x4*)(Bt + (size_t)col * ldb + k0);
      bq[nt].q[0] = p[lh];      // K = k0 + lh*8 + {0..7}
      bq[nt].q[1] = p[2 + lh];  // K = k0 + 16 + lh*8 + {0..7}
    }
#pragma unroll
    for (int mt = 0; mt < 2; mt++)
#pragma unroll
      for (int nt = 0; nt < 2; nt++)
        acc[mt][nt] = __builtin_amdgcn_wmma_f32_16x16x32_bf16(
            false, a[mt].v, false, bq[nt].v, (short)0, acc[mt][nt], false, false);
  }

#pragma unroll
  for (int mt = 0; mt < 2; mt++)
#pragma unroll
    for (int nt = 0; nt < 2; nt++) {
      int col = n0 + nt * 16 + lm;
      if (col >= N) continue;
      float bv = bias ? bias[col] : 0.f;
#pragma unroll
      for (int i = 0; i < 8; i++) {
        int row = m0 + mt * 16 + lh * 8 + i;
        if (row >= M) continue;
        float v = acc[mt][nt][i] + bv;
        if (act == 1) v = gelu_tanh(v);
        long long idx = coff + (long long)row * ldc + col;
        if (C)   C[idx] = v;
        if (Obf) Obf[idx] = f2bf(v);
      }
    }
}

// ---------------- embedding + LN ----------------
__global__ __launch_bounds__(256)
void embed_ln_kernel(const int* __restrict__ x, const float* __restrict__ we,
                     const float* __restrict__ pe, const float* __restrict__ te,
                     const float* __restrict__ w, const float* __restrict__ b,
                     float* __restrict__ h32, unsigned short* __restrict__ h16)
{
  int row = blockIdx.x;         // 0..2047
  int s   = row & (SS - 1);
  int tid = threadIdx.x;
  int wid = x[row];
  float vals[3], lsum = 0.f, lsq = 0.f;
#pragma unroll
  for (int i = 0; i < 3; i++) {
    int j = tid + i * 256;
    float v = we[(size_t)wid * HH + j] + pe[(size_t)s * HH + j] + te[j];
    vals[i] = v; lsum += v; lsq += v * v;
  }
  __shared__ float s1[256], s2[256];
  s1[tid] = lsum; s2[tid] = lsq; __syncthreads();
  for (int st = 128; st > 0; st >>= 1) {
    if (tid < st) { s1[tid] += s1[tid + st]; s2[tid] += s2[tid + st]; }
    __syncthreads();
  }
  float mean = s1[0] * (1.0f / HH);
  float var  = s2[0] * (1.0f / HH) - mean * mean;
  float inv  = rsqrtf(var + 1e-12f);
#pragma unroll
  for (int i = 0; i < 3; i++) {
    int j = tid + i * 256;
    float v = (vals[i] - mean) * inv * w[j] + b[j];
    h32[(size_t)row * HH + j] = v;
    h16[(size_t)row * HH + j] = f2bf(v);
  }
}

// ---------------- residual + LN ----------------
__global__ __launch_bounds__(256)
void residual_ln_kernel(const float* __restrict__ in, const float* __restrict__ res,
                        const float* __restrict__ w, const float* __restrict__ b,
                        float* __restrict__ h32, unsigned short* __restrict__ h16)
{
  int row = blockIdx.x;
  int tid = threadIdx.x;
  float vals[3], lsum = 0.f, lsq = 0.f;
#pragma unroll
  for (int i = 0; i < 3; i++) {
    int j = tid + i * 256;
    float v = in[(size_t)row * HH + j] + res[(size_t)row * HH + j];
    vals[i] = v; lsum += v; lsq += v * v;
  }
  __shared__ float s1[256], s2[256];
  s1[tid] = lsum; s2[tid] = lsq; __syncthreads();
  for (int st = 128; st > 0; st >>= 1) {
    if (tid < st) { s1[tid] += s1[tid + st]; s2[tid] += s2[tid + st]; }
    __syncthreads();
  }
  float mean = s1[0] * (1.0f / HH);
  float var  = s2[0] * (1.0f / HH) - mean * mean;
  float inv  = rsqrtf(var + 1e-12f);
#pragma unroll
  for (int i = 0; i < 3; i++) {
    int j = tid + i * 256;
    float v = (vals[i] - mean) * inv * w[j] + b[j];
    h32[(size_t)row * HH + j] = v;
    h16[(size_t)row * HH + j] = f2bf(v);
  }
}

// ---------------- row softmax (scale folded in), f32 -> bf16 ----------------
__global__ __launch_bounds__(256)
void softmax_kernel(const float* __restrict__ sc, unsigned short* __restrict__ p, float scale)
{
  size_t row = blockIdx.x;
  int tid = threadIdx.x;
  float v = sc[row * 256 + tid] * scale;
  __shared__ float sm[256];
  sm[tid] = v; __syncthreads();
  for (int st = 128; st > 0; st >>= 1) {
    if (tid < st) sm[tid] = fmaxf(sm[tid], sm[tid + st]);
    __syncthreads();
  }
  float mx = sm[0]; __syncthreads();
  float e = __expf(v - mx);
  sm[tid] = e; __syncthreads();
  for (int st = 128; st > 0; st >>= 1) {
    if (tid < st) sm[tid] += sm[tid + st];
    __syncthreads();
  }
  float s = sm[0];
  p[row * 256 + tid] = f2bf(e / s);
}

// ---------------- V transpose: v16[(b*256+s)*768 + h*64+d] -> vt[((b*12+h)*64+d)*256+s] ----------------
__global__ __launch_bounds__(256)
void vtrans_kernel(const unsigned short* __restrict__ v, unsigned short* __restrict__ vt)
{
  int idx = blockIdx.x * 256 + threadIdx.x;
  if (idx >= MTOK * HH) return;
  int s  = idx & 255;
  int d  = (idx >> 8) & 63;
  int bh = idx >> 14;
  int b  = bh / NHEAD, h = bh % NHEAD;
  vt[idx] = v[((size_t)(b * 256 + s)) * HH + h * 64 + d];
}

// ---------------- CRF per-batch log-likelihood ----------------
__global__ __launch_bounds__(32)
void crf_kernel(const float* __restrict__ em, const int* __restrict__ tgt,
                const float* __restrict__ cs, const float* __restrict__ ce,
                const float* __restrict__ ctr, float* __restrict__ perb)
{
  int b = blockIdx.x, lane = threadIdx.x;
  __shared__ float tr[NC * NC], alpha[NC], nxt[NC];
  for (int i = lane; i < NC * NC; i += 32) tr[i] = ctr[i];
  if (lane < NC) alpha[lane] = cs[lane] + em[((size_t)b * SS + 0) * NC + lane];
  __syncthreads();
  for (int s = 1; s < SS; s++) {
    float old[NC];
#pragma unroll
    for (int i = 0; i < NC; i++) old[i] = alpha[i];
    int tc = tgt[b * SS + s];
    float m = (tc > -1) ? 1.f : 0.f;
    if (lane < NC) {
      float mx = -1e30f;
#pragma unroll
      for (int i = 0; i < NC; i++) mx = fmaxf(mx, old[i] + tr[i * NC + lane]);
      float ssum = 0.f;
#pragma unroll
      for (int i = 0; i < NC; i++) ssum += __expf(old[i] + tr[i * NC + lane] - mx);
      float v = mx + __logf(ssum) + em[((size_t)b * SS + s) * NC + lane];
      nxt[lane] = (m > 0.f) ? v : old[lane];
    }
    __syncthreads();
    if (lane < NC) alpha[lane] = nxt[lane];
    __syncthreads();
  }
  if (lane == 0) {
    float mx = -1e30f;
    for (int j = 0; j < NC; j++) mx = fmaxf(mx, alpha[j] + ce[j]);
    float ssum = 0.f;
    for (int j = 0; j < NC; j++) ssum += __expf(alpha[j] + ce[j] - mx);
    float den = mx + __logf(ssum);

    int t0 = tgt[b * SS];
    float num = cs[t0] + em[((size_t)b * SS) * NC + t0];
    int msum = 0;
    for (int s = 0; s < SS; s++) msum += (tgt[b * SS + s] > -1) ? 1 : 0;
    int seqend = msum - 1;
    for (int s = 1; s < SS; s++) {
      int tp = tgt[b * SS + s - 1], tc = tgt[b * SS + s];
      float m = (tc > -1) ? 1.f : 0.f;
      num += m * (tr[tp * NC + tc] + em[((size_t)b * SS + s) * NC + tc]);
    }
    num += ce[tgt[b * SS + seqend]];
    perb[b] = num - den;
  }
}

__global__ void loss_kernel(const float* __restrict__ perb, float* __restrict__ out)
{
  if (threadIdx.x == 0 && blockIdx.x == 0) {
    float s = 0.f;
    for (int i = 0; i < BB; i++) s += perb[i];
    out[0] = -s / (float)BB;
  }
}

// ---------------- host side ----------------
static void gemm(hipStream_t st,
                 const void* A, int lda, long long sAb, long long sAh,
                 const void* Bt, int ldb, long long sBb, long long sBh,
                 void* C, void* Obf, int ldc, long long sCb, long long sCh,
                 const void* bias, int M, int N, int K, int batch, int bh, int act)
{
  dim3 g((M + 63) / 64, (N + 63) / 64, batch);
  wmma_gemm_bf16<<<g, 128, 0, st>>>(
      (const unsigned short*)A, lda, sAb, sAh,
      (const unsigned short*)Bt, ldb, sBb, sBh,
      (float*)C, (unsigned short*)Obf, ldc, sCb, sCh,
      (const float*)bias, M, N, K, bh, act);
}

extern "C" void kernel_launch(void* const* d_in, const int* in_sizes, int n_in,
                              void* d_out, int out_size, void* d_ws, size_t ws_size,
                              hipStream_t stream)
{
  (void)in_sizes; (void)n_in; (void)out_size; (void)ws_size;
  const int*   x          = (const int*)d_in[0];
  const int*   target     = (const int*)d_in[1];
  const float* word_emb   = (const float*)d_in[2];
  const float* pos_emb    = (const float*)d_in[3];
  const float* type_emb   = (const float*)d_in[4];
  const float* emb_ln_w   = (const float*)d_in[5];
  const float* emb_ln_b   = (const float*)d_in[6];
  const float* qkv_w      = (const float*)d_in[7];
  const float* qkv_b      = (const float*)d_in[8];
  const float* attn_out_w = (const float*)d_in[9];
  const float* attn_out_b = (const float*)d_in[10];
  const float* attn_ln_w  = (const float*)d_in[11];
  const float* attn_ln_b  = (const float*)d_in[12];
  const float* ffn_w1     = (const float*)d_in[13];
  const float* ffn_b1     = (const float*)d_in[14];
  const float* ffn_w2     = (const float*)d_in[15];
  const float* ffn_b2     = (const float*)d_in[16];
  const float* ffn_ln_w   = (const float*)d_in[17];
  const float* ffn_ln_b   = (const float*)d_in[18];
  const float* cls_w      = (const float*)d_in[19];
  const float* cls_b      = (const float*)d_in[20];
  const float* crf_start  = (const float*)d_in[21];
  const float* crf_end    = (const float*)d_in[22];
  const float* crf_trans  = (const float*)d_in[23];

  char* base = (char*)d_ws;
  size_t off = 0;
  auto carve = [&](size_t bytes) -> void* {
    off = (off + 255) & ~(size_t)255;
    void* p = base + off;
    off += bytes;
    return p;
  };

  const size_t HHm = (size_t)HH * HH;                  // 589824
  unsigned short* qkvT = (unsigned short*)carve(NLAY * 3 * HHm * 2);
  unsigned short* aoT  = (unsigned short*)carve(NLAY * HHm * 2);
  unsigned short* f1T  = (unsigned short*)carve((size_t)NLAY * HH * FFD * 2);
  unsigned short* f2T  = (unsigned short*)carve((size_t)NLAY * FFD * HH * 2);
  unsigned short* clsT = (unsigned short*)carve((size_t)NC * HH * 2);

  float*          h32  = (float*)carve((size_t)MTOK * HH * 4);
  unsigned short* h16  = (unsigned short*)carve((size_t)MTOK * HH * 2);
  unsigned short* q16  = (unsigned short*)carve((size_t)MTOK * HH * 2);
  unsigned short* k16  = (unsigned short*)carve((size_t)MTOK * HH * 2);
  unsigned short* v16  = (unsigned short*)carve((size_t)MTOK * HH * 2);
  unsigned short* vt16 = (unsigned short*)carve((size_t)MTOK * HH * 2);
  float*          sc32 = (float*)carve((size_t)BB * NHEAD * SS * SS * 4);
  unsigned short* p16  = (unsigned short*)carve((size_t)BB * NHEAD * SS * SS * 2);
  unsigned short* ctx16= (unsigned short*)carve((size_t)MTOK * HH * 2);
  unsigned short* f116 = (unsigned short*)carve((size_t)MTOK * FFD * 2);
  float*          tmp32= (float*)carve((size_t)MTOK * HH * 4);
  float*          em32 = (float*)carve((size_t)MTOK * NC * 4);
  float*          perb = (float*)carve(BB * 4);

  // ---- weight conversion (f32 -> transposed bf16) ----
  {
    dim3 g1((HHm + 255) / 256, NLAY * 3);
    convT_kernel<<<g1, 256, 0, stream>>>(qkv_w, qkvT, HH, HH);
    dim3 g2((HHm + 255) / 256, NLAY);
    convT_kernel<<<g2, 256, 0, stream>>>(attn_out_w, aoT, HH, HH);
    dim3 g3(((size_t)HH * FFD + 255) / 256, NLAY);
    convT_kernel<<<g3, 256, 0, stream>>>(ffn_w1, f1T, HH, FFD);
    convT_kernel<<<g3, 256, 0, stream>>>(ffn_w2, f2T, FFD, HH);
    dim3 g4(((size_t)HH * NC + 255) / 256, 1);
    convT_kernel<<<g4, 256, 0, stream>>>(cls_w, clsT, HH, NC);
  }

  // ---- embedding + LN ----
  embed_ln_kernel<<<MTOK, 256, 0, stream>>>(x, word_emb, pos_emb, type_emb,
                                            emb_ln_w, emb_ln_b, h32, h16);

  const long long sTok = (long long)SS * HH;   // 196608
  const long long sSc  = (long long)SS * SS;   // 65536
  const long long sVt  = (long long)DHD * SS;  // 16384

  for (int l = 0; l < NLAY; l++) {
    const unsigned short* qkvTl = qkvT + (size_t)l * 3 * HHm;
    // Q, K, V projections (bf16 outputs)
    gemm(stream, h16, HH, 0, 0, qkvTl + 0 * HHm, HH, 0, 0,
         nullptr, q16, HH, 0, 0, qkv_b + (size_t)(l * 3 + 0) * HH,
         MTOK, HH, HH, 1, 1, 0);
    gemm(stream, h16, HH, 0, 0, qkvTl + 1 * HHm, HH, 0, 0,
         nullptr, k16, HH, 0, 0, qkv_b + (size_t)(l * 3 + 1) * HH,
         MTOK, HH, HH, 1, 1, 0);
    gemm(stream, h16, HH, 0, 0, qkvTl + 2 * HHm, HH, 0, 0,
         nullptr, v16, HH, 0, 0, qkv_b + (size_t)(l * 3 + 2) * HH,
         MTOK, HH, HH, 1, 1, 0);

    // scores[b,h] = Q_bh (256x64) * K_bh^T  -> f32
    gemm(stream, q16, HH, sTok, DHD, k16, HH, sTok, DHD,
         sc32, nullptr, SS, (long long)NHEAD * sSc, sSc, nullptr,
         SS, SS, DHD, BB * NHEAD, NHEAD, 0);

    // softmax (scale = 1/sqrt(64)) -> bf16 P
    softmax_kernel<<<BB * NHEAD * SS, 256, 0, stream>>>(sc32, p16, 0.125f);

    // V transpose for context GEMM
    vtrans_kernel<<<(MTOK * HH + 255) / 256, 256, 0, stream>>>(v16, vt16);

    // ctx[b,h] = P (256x256) * V_bh (256x64) -> bf16, laid out [b*256+s][h*64+d]
    gemm(stream, p16, SS, (long long)NHEAD * sSc, sSc,
         vt16, SS, (long long)NHEAD * sVt, sVt,
         nullptr, ctx16, HH, sTok, DHD, nullptr,
         SS, DHD, SS, BB * NHEAD, NHEAD, 0);

    // attention output projection -> f32 tmp
    gemm(stream, ctx16, HH, 0, 0, aoT + (size_t)l * HHm, HH, 0, 0,
         tmp32, nullptr, HH, 0, 0, attn_out_b + (size_t)l * HH,
         MTOK, HH, HH, 1, 1, 0);

    residual_ln_kernel<<<MTOK, 256, 0, stream>>>(tmp32, h32,
        attn_ln_w + (size_t)l * HH, attn_ln_b + (size_t)l * HH, h32, h16);

    // FFN1 + GELU -> bf16
    gemm(stream, h16, HH, 0, 0, f1T + (size_t)l * HH * FFD, HH, 0, 0,
         nullptr, f116, FFD, 0, 0, ffn_b1 + (size_t)l * FFD,
         MTOK, FFD, HH, 1, 1, 1);

    // FFN2 -> f32 tmp
    gemm(stream, f116, FFD, 0, 0, f2T + (size_t)l * FFD * HH, FFD, 0, 0,
         tmp32, nullptr, HH, 0, 0, ffn_b2 + (size_t)l * HH,
         MTOK, HH, FFD, 1, 1, 0);

    residual_ln_kernel<<<MTOK, 256, 0, stream>>>(tmp32, h32,
        ffn_ln_w + (size_t)l * HH, ffn_ln_b + (size_t)l * HH, h32, h16);
  }

  // classifier -> emissions f32 [2048 x 9]
  gemm(stream, h16, HH, 0, 0, clsT, HH, 0, 0,
       em32, nullptr, NC, 0, 0, cls_b, MTOK, NC, HH, 1, 1, 0);

  // CRF loss
  crf_kernel<<<BB, 32, 0, stream>>>(em32, target, crf_start, crf_end, crf_trans, perb);
  loss_kernel<<<1, 1, 0, stream>>>(perb, (float*)d_out);
}
